// DifferentiableVoronoiPropagation_86981677678607
// MI455X (gfx1250) — matrix operand order repaired and checked by reading, not compiled, not run on hardware
//
#include <hip/hip_runtime.h>
#include <hip/hip_bf16.h>
#include <math.h>

// ---------------------------------------------------------------------------
// DifferentiableVoronoiPropagation for MI455X (gfx1250, wave32, WMMA)
//   conv2 (89 GFLOP) -> LDS-staged implicit GEMM on v_wmma_f32_16x16x32_f16.
//   * K reordered tap-major (K' = tap*64 + ic): fixed tap per 32-wide K-step.
//   * f16 activations stored with a zero halo (228x228): branchless A-staging.
//   * B tile staged by the Tensor Data Mover (tensor_load_to_lds + TENSORcnt),
//     with D# LDS padding reproducing the 80-B conflict-free row stride.
//   * Double-buffered LDS tiles: one barrier per K-step, stage/compute overlap.
// ---------------------------------------------------------------------------

#define HH 224
#define WW 224
#define HWP (HH * WW)          // 50176
#define BB 8
#define NPIX (BB * HWP)        // 401408
#define C1 64
#define C2 192
#define KDIM 576               // C1 * 9
#define KSTEP 32
#define BM 64                  // pixels per block
#define LSTR 40                // padded LDS row stride (halves) = 80 B
#define WPAD 228               // 224 + 2*2 halo
#define PLANE (WPAD * WPAD)    // padded plane (51984)
#define NCENT 196
#define NB_ 10
#define SLC 20

typedef _Float16 half16 __attribute__((ext_vector_type(16)));
typedef _Float16 half8  __attribute__((ext_vector_type(8)));
typedef float    float8 __attribute__((ext_vector_type(8)));
typedef unsigned int uint4v __attribute__((ext_vector_type(4)));
typedef int int4v  __attribute__((ext_vector_type(4)));
typedef int int8v  __attribute__((ext_vector_type(8)));

#if defined(__has_builtin)
#if __has_builtin(__builtin_amdgcn_tensor_load_to_lds) && \
    __has_builtin(__builtin_amdgcn_s_wait_tensorcnt)
#define USE_TDM 1
#endif
#endif

#ifdef USE_TDM
// TDM: DMA a [192 rows x 32 halves] tile (row stride KDIM halves) from wt
// into LDS with +16 B pad every 64 B -> 80 B LDS row stride (== LSTR*2).
__device__ __forceinline__ void tdm_load_B(const _Float16* gsrc,
                                           unsigned ldsOff) {
  unsigned long long ga = (unsigned long long)(size_t)(const void*)gsrc;
  uint4v g0;
  g0[0] = 1u;                                        // count=1, user mode
  g0[1] = ldsOff;                                    // lds_addr (bytes)
  g0[2] = (unsigned)(ga & 0xFFFFFFFFull);            // global_addr[31:0]
  g0[3] = (unsigned)((ga >> 32) & 0x01FFFFFFull)     // global_addr[56:32]
          | (2u << 30);                              // type=2 ("image")
  int8v g1;
  g1[0] = (1 << 16)                                  // data_size = 2 B
        | (1 << 20)                                  // pad_enable
        | (3 << 22)                                  // pad_interval: 16 DW (64 B)
        | (3 << 25);                                 // pad_amount: 4 DW (16 B)
  g1[1] = (int)(576u << 16);                         // tensor_dim0 = 576 (lo16)
  g1[2] = (int)(192u << 16);                         // dim0 hi=0 | tensor_dim1=192
  g1[3] = (int)(32u << 16);                          // dim1 hi=0 | tile_dim0=32
  g1[4] = 192;                                       // tile_dim1=192, tile_dim2=0
  g1[5] = 576;                                       // tensor_dim0_stride lo32
  g1[6] = 0;                                         // stride hi | dim1_stride lo
  g1[7] = 0;
  int4v gz = {0, 0, 0, 0};
#if __clang_major__ >= 23
  int8v gz8 = {0, 0, 0, 0, 0, 0, 0, 0};
  __builtin_amdgcn_tensor_load_to_lds(g0, g1, gz, gz, gz8, 0);
#else
  __builtin_amdgcn_tensor_load_to_lds(g0, g1, gz, gz, 0);
#endif
}
#endif

// --------------------------- gray + sobel ----------------------------------
__global__ __launch_bounds__(256) void k_sobel(const float* __restrict__ x,
                                               float* __restrict__ grad) {
  int p = blockIdx.x * 256 + threadIdx.x;
  if (p >= NPIX) return;
  int n = p / HWP, pi = p % HWP, y = pi / WW, xc = pi % WW;
  const float* xb = x + (size_t)n * 3 * HWP;
  auto gray = [&](int yy, int xx) -> float {
    if (yy < 0 || yy >= HH || xx < 0 || xx >= WW) return 0.f;
    size_t o = (size_t)yy * WW + xx;
    return 0.299f * xb[o] + 0.587f * xb[o + HWP] + 0.114f * xb[o + 2 * HWP];
  };
  float g00 = gray(y - 1, xc - 1), g01 = gray(y - 1, xc), g02 = gray(y - 1, xc + 1);
  float g10 = gray(y,     xc - 1),                        g12 = gray(y,     xc + 1);
  float g20 = gray(y + 1, xc - 1), g21 = gray(y + 1, xc), g22 = gray(y + 1, xc + 1);
  float gx = -g00 + g02 - 2.f * g10 + 2.f * g12 - g20 + g22;
  float gy = -g00 - 2.f * g01 - g02 + g20 + 2.f * g21 + g22;
  grad[p] = sqrtf(gx * gx + gy * gy);
}

// --------------------------- conv1 (3->64, 3x3) ----------------------------
__global__ __launch_bounds__(256) void k_conv1(const float* __restrict__ x,
                                               const float* __restrict__ w1,
                                               const float* __restrict__ b1,
                                               float* __restrict__ hpre) {
  __shared__ float wS[C1 * 27];
  __shared__ float bS[C1];
  for (int i = threadIdx.x; i < C1 * 27; i += 256) wS[i] = w1[i];
  for (int i = threadIdx.x; i < C1; i += 256) bS[i] = b1[i];
  __syncthreads();
  int p = blockIdx.x * 256 + threadIdx.x;
  if (p >= NPIX) return;
  int n = p / HWP, pi = p % HWP, y = pi / WW, xc = pi % WW;
  float in[27];
#pragma unroll
  for (int c = 0; c < 3; ++c)
#pragma unroll
    for (int dy = -1; dy <= 1; ++dy)
#pragma unroll
      for (int dx = -1; dx <= 1; ++dx) {
        int yy = y + dy, xx = xc + dx;
        float v = 0.f;
        if (yy >= 0 && yy < HH && xx >= 0 && xx < WW)
          v = x[((size_t)(n * 3 + c) * HWP) + (size_t)yy * WW + xx];
        in[c * 9 + (dy + 1) * 3 + (dx + 1)] = v;
      }
  for (int oc = 0; oc < C1; ++oc) {
    float acc = bS[oc];
#pragma unroll
    for (int t = 0; t < 27; ++t) acc = fmaf(wS[oc * 27 + t], in[t], acc);
    hpre[((size_t)(n * C1 + oc) * HWP) + pi] = acc;
  }
}

// --------------------- per-channel BN stats (sum, sumsq) -------------------
__global__ __launch_bounds__(256) void k_bn_stats(const float* __restrict__ src,
                                                  float* __restrict__ stats, int C) {
  int c = blockIdx.y;
  __shared__ float s1[256], s2[256];
  float a = 0.f, b = 0.f;
  size_t per = (size_t)BB * HWP;
  for (size_t e = (size_t)blockIdx.x * 256 + threadIdx.x; e < per;
       e += (size_t)gridDim.x * 256) {
    int n = (int)(e / HWP), pi = (int)(e % HWP);
    float v = src[((size_t)(n * C + c) * HWP) + pi];
    a += v; b += v * v;
  }
  s1[threadIdx.x] = a; s2[threadIdx.x] = b;
  __syncthreads();
  for (int s = 128; s > 0; s >>= 1) {
    if ((int)threadIdx.x < s) { s1[threadIdx.x] += s1[threadIdx.x + s];
                                s2[threadIdx.x] += s2[threadIdx.x + s]; }
    __syncthreads();
  }
  if (threadIdx.x == 0) { atomicAdd(&stats[2 * c], s1[0]);
                          atomicAdd(&stats[2 * c + 1], s2[0]); }
}

// --------- BN+ReLU on conv1 output -> f16 with 2-px zero halo --------------
__global__ __launch_bounds__(256) void k_bn1_f16(const float* __restrict__ hpre,
                                                 const float* __restrict__ stats,
                                                 const float* __restrict__ g,
                                                 const float* __restrict__ be,
                                                 _Float16* __restrict__ hf) {
  size_t i = (size_t)blockIdx.x * 256 + threadIdx.x;
  if (i >= (size_t)NPIX * C1) return;
  int pi = (int)(i % HWP);
  size_t nc = i / HWP;                      // n*C1 + c
  int c = (int)(nc % C1);
  int y = pi / WW, xx = pi % WW;
  float cnt = (float)((size_t)BB * HWP);
  float m = stats[2 * c] / cnt;
  float var = stats[2 * c + 1] / cnt - m * m;
  float v = (hpre[i] - m) * (g[c] * rsqrtf(var + 1e-5f)) + be[c];
  hf[nc * PLANE + (size_t)(y + 2) * WPAD + (xx + 2)] = (_Float16)fmaxf(v, 0.f);
}

// -------- w2 f32 OIHW -> f16 [oc][K'] with K' = tap*64 + ic ----------------
__global__ __launch_bounds__(256) void k_w2_f16(const float* __restrict__ w2,
                                                _Float16* __restrict__ wt) {
  int i = blockIdx.x * 256 + threadIdx.x;
  if (i >= C2 * KDIM) return;
  int oc = i / KDIM, r = i - oc * KDIM;   // r = ic*9 + tap
  int ic = r / 9, tap = r - ic * 9;
  wt[(size_t)oc * KDIM + tap * 64 + ic] = (_Float16)w2[i];
}

// ---------- conv2 implicit GEMM with LDS staging + WMMA --------------------
// Block: 256 thr (8 waves).  Tile: M=64 pixels x N=192 oc.  K' loop: 9 taps x 2.
// Wave (msub = w&3, ng = w>>2): 16 pixels x 96 channels -> 6 f32 16x16 accs.
__global__ __launch_bounds__(256) void k_conv2_wmma(const _Float16* __restrict__ hf,
                                                    const _Float16* __restrict__ wt,
                                                    float* __restrict__ feats) {
  __shared__ alignas(16) _Float16 ldsA[2][BM * LSTR];   // [m][kk pad40]  2x5 KB
  __shared__ alignas(16) _Float16 ldsB[2][C2 * LSTR];   // [oc][kk pad40] 2x15 KB

  const int pix0 = blockIdx.x * BM;      // 784 tiles per image: never straddles
  const int n = pix0 / HWP;
  const int pimg0 = pix0 - n * HWP;
  const _Float16* hb = hf + (size_t)n * C1 * PLANE;

  // A-staging role: fixed pixel per thread, 8 channel slots per step
  const int mA = threadIdx.x & 63;
  const int kkA = threadIdx.x >> 6;      // 0..3, step 4
  const int pA = pimg0 + mA;
  const int pyA = pA / WW, pxA = pA % WW;

  // compute role
  const int lane = threadIdx.x & 31;
  const int wave = threadIdx.x >> 5;
  const int msub = wave & 3;             // 16-pixel subtile
  const int ng = wave >> 2;              // 0/1 -> channel half (96 each)
  const int grp = lane >> 4;
  const int ml = lane & 15;

  union Frag { half16 v; half8 h[2]; };
  float8 acc[6];
#pragma unroll
  for (int j = 0; j < 6; ++j) acc[j] = float8{0.f,0.f,0.f,0.f,0.f,0.f,0.f,0.f};

  for (int ky = 0; ky < 3; ++ky) {
    for (int kx = 0; kx < 3; ++kx) {
      // halo-padded source: always in-bounds, halo is zero
      const _Float16* srcA =
          hb + (size_t)(pyA + 2 + 2 * (ky - 1)) * WPAD + (pxA + 2 + 2 * (kx - 1));
      const int tap = ky * 3 + kx;
      for (int half = 0; half < 2; ++half) {
        const int st = tap * 2 + half;
        const int buf = st & 1;
        const int k0 = tap * 64 + half * 32;     // K' base (fixed tap)
        const int icB = half * 32;

        // ---- stage B: TDM DMA (wave 0) or vector-copy fallback ----
#ifdef USE_TDM
        if (threadIdx.x < 32)
          tdm_load_B(wt + k0, (unsigned)(size_t)(void*)&ldsB[buf][0]);
#else
#pragma unroll
        for (int i = 0; i < 3; ++i) {
          int e8 = threadIdx.x + i * 256;        // 768 8-half chunks
          int oc = e8 >> 2;
          int kc = (e8 & 3) * 8;
          *(half8*)&ldsB[buf][oc * LSTR + kc] =
              *(const half8*)&wt[(size_t)oc * KDIM + k0 + kc];
        }
#endif
        // ---- stage A: 8 unconditional channel-strided loads ----
#pragma unroll
        for (int i = 0; i < 8; ++i) {
          int kk = kkA + i * 4;
          ldsA[buf][mA * LSTR + kk] = srcA[(size_t)(icB + kk) * PLANE];
        }
        if (k0 + KSTEP < KDIM)                   // global_prefetch_b8
          __builtin_prefetch(wt + (size_t)(k0 + KSTEP), 0, 1);
#ifdef USE_TDM
        if (threadIdx.x < 32) __builtin_amdgcn_s_wait_tensorcnt(0);
#endif
        __syncthreads();   // publish stage(st); compute(st-1) ordered earlier

        // ---- fragments + 6 WMMAs per wave ----
        // A 16x32: lane grp g, row ml: halves 0-7 -> kk g*8+0..7,
        // halves 8-15 -> kk 16+g*8+0..7 (ISA 7.12.2).
        Frag a;
        a.h[0] = *(const half8*)&ldsA[buf][(msub * 16 + ml) * LSTR + grp * 8];
        a.h[1] = *(const half8*)&ldsA[buf][(msub * 16 + ml) * LSTR + 16 + grp * 8];
#pragma unroll
        for (int j = 0; j < 6; ++j) {
          int oc = ng * 96 + j * 16 + ml;        // B col N = ml
          Frag b;                                // same kk order as A
          b.h[0] = *(const half8*)&ldsB[buf][oc * LSTR + grp * 8];
          b.h[1] = *(const half8*)&ldsB[buf][oc * LSTR + 16 + grp * 8];
          acc[j] = __builtin_amdgcn_wmma_f32_16x16x32_f16(
              false, a.v, false, b.v, (short)0, acc[j], false, false);
        }
      }
    }
  }

  // C/D 16x16 f32: VGPR r -> M = r + 8*grp, N = lane%16 (32B run per lane)
#pragma unroll
  for (int j = 0; j < 6; ++j) {
    int oc = ng * 96 + j * 16 + ml;
    float* outc = feats + (size_t)(n * C2 + oc) * HWP + pimg0 + msub * 16;
#pragma unroll
    for (int r = 0; r < 8; ++r)
      outc[r + (grp << 3)] = acc[j][r];
  }
}

// ----------- BN+ReLU on feats (in place) + channel mean --------------------
__global__ __launch_bounds__(256) void k_bn2_mean(float* __restrict__ feats,
                                                  const float* __restrict__ stats,
                                                  const float* __restrict__ g,
                                                  const float* __restrict__ be,
                                                  float* __restrict__ mf) {
  __shared__ float sm[C2], ss[C2];
  float cnt = (float)((size_t)BB * HWP);
  for (int c = threadIdx.x; c < C2; c += 256) {
    float m = stats[2 * c] / cnt;
    float var = stats[2 * c + 1] / cnt - m * m;
    sm[c] = m;
    ss[c] = g[c] * rsqrtf(var + 1e-5f);
  }
  __syncthreads();
  int p = blockIdx.x * 256 + threadIdx.x;
  if (p >= NPIX) return;
  int n = p / HWP, pi = p % HWP;
  float accv = 0.f;
  for (int c = 0; c < C2; ++c) {
    size_t idx = ((size_t)(n * C2 + c) * HWP) + pi;
    float v = (feats[idx] - sm[c]) * ss[c] + be[c];
    v = fmaxf(v, 0.f);
    feats[idx] = v;
    accv += v;
  }
  mf[p] = accv / (float)C2;
}

// ---------------- per-direction penalty maps (wrap-around) -----------------
__global__ __launch_bounds__(256) void k_pens(const float* __restrict__ x,
                                              const float* __restrict__ grad,
                                              const float* __restrict__ mf,
                                              float* __restrict__ pens) {
  int p = blockIdx.x * 256 + threadIdx.x;
  if (p >= NPIX) return;
  int n = p / HWP, pi = p % HWP, y = pi / WW, xc = pi % WW;
  float gv = grad[p];
  float g2 = gv * gv;
  float wg = g2 * g2 * 10.f;                 // grad^4 * gradient_weight
  const int dys[4] = {-1, 1, 0, 0}, dxs[4] = {0, 0, -1, 1};
  for (int d = 0; d < 4; ++d) {
    int yy = (y - dys[d] + HH) % HH;
    int xx = (xc - dxs[d] + WW) % WW;
    float cd = 0.f;
    for (int c = 0; c < 3; ++c) {
      const float* xb = x + (size_t)(n * 3 + c) * HWP;
      cd += fabsf(xb[(size_t)y * WW + xc] - xb[(size_t)yy * WW + xx]);
    }
    const float* mb = mf + (size_t)n * HWP;
    float fd = fabsf(mb[(size_t)y * WW + xc] - mb[(size_t)yy * WW + xx]);
    pens[(size_t)d * NPIX + p] = wg + cd * 10.f + fd * 5.f;
  }
}

// --------- sequential collision-avoiding minima scan (1 block/image) -------
__global__ __launch_bounds__(256) void k_minima(const float* __restrict__ grad,
                                                int* __restrict__ cents) {
  int n = blockIdx.x;
  __shared__ unsigned occ[(HWP + 31) / 32];   // 1568 words
  __shared__ float smin[256];
  __shared__ int sidx[256];
  for (int i = threadIdx.x; i < (HWP + 31) / 32; i += 256) occ[i] = 0u;
  __syncthreads();
  const float* gb = grad + (size_t)n * HWP;
  for (int ci = 0; ci < NCENT; ++ci) {
    int iy = ci / 14, jx = ci % 14;
    int y = 16 * iy + 8, x = 16 * jx + 8;     // grid centroids (gy=gx=16)
    int sy = y - NB_; if (sy < 0) sy = 0; if (sy > HH - SLC) sy = HH - SLC;
    int sx = x - NB_; if (sx < 0) sx = 0; if (sx > WW - SLC) sx = WW - SLC;
    float mv = INFINITY;
    for (int q = threadIdx.x; q < SLC * SLC; q += 256) {
      int r = q / SLC, c = q % SLC;
      int row = sy + r, col = sx + c;
      bool valid = (row >= y - NB_) && (row < y + NB_) &&
                   (col >= x - NB_) && (col < x + NB_);
      if (valid) mv = fminf(mv, gb[(size_t)row * WW + col]);
    }
    smin[threadIdx.x] = mv;
    __syncthreads();
    for (int s = 128; s > 0; s >>= 1) {
      if ((int)threadIdx.x < s)
        smin[threadIdx.x] = fminf(smin[threadIdx.x], smin[threadIdx.x + s]);
      __syncthreads();
    }
    mv = smin[0];
    int best = 1 << 30;
    for (int q = threadIdx.x; q < SLC * SLC; q += 256) {
      int r = q / SLC, c = q % SLC;
      int row = sy + r, col = sx + c;
      bool valid = (row >= y - NB_) && (row < y + NB_) &&
                   (col >= x - NB_) && (col < x + NB_);
      if (valid) {
        float v = gb[(size_t)row * WW + col];
        int cell = row * WW + col;
        bool occd = (occ[cell >> 5] >> (cell & 31)) & 1u;
        if (v == mv && !occd) best = min(best, q);
      }
    }
    sidx[threadIdx.x] = best;
    __syncthreads();
    for (int s = 128; s > 0; s >>= 1) {
      if ((int)threadIdx.x < s)
        sidx[threadIdx.x] = min(sidx[threadIdx.x], sidx[threadIdx.x + s]);
      __syncthreads();
    }
    if (threadIdx.x == 0) {
      int fi = sidx[0];
      int ny = y, nx = x;
      if (fi < (1 << 30)) {
        ny = sy + fi / SLC; nx = sx + fi % SLC;
        int cell = ny * WW + nx;
        occ[cell >> 5] |= (1u << (cell & 31));
      }
      cents[(n * NCENT + ci) * 2] = ny;
      cents[(n * NCENT + ci) * 2 + 1] = nx;
    }
    __syncthreads();
  }
}

// ------------------------- propagation setup -------------------------------
__global__ __launch_bounds__(256) void k_init_dist(float* __restrict__ dist,
                                                   int* __restrict__ mask) {
  int p = blockIdx.x * 256 + threadIdx.x;
  if (p >= NPIX) return;
  dist[p] = INFINITY;
  mask[p] = -1;
}

__global__ __launch_bounds__(256) void k_seed(const int* __restrict__ cents,
                                              float* __restrict__ dist,
                                              int* __restrict__ mask) {
  int t = blockIdx.x * 256 + threadIdx.x;
  if (t >= BB * NCENT) return;
  int n = t / NCENT, ci = t % NCENT;
  int y = cents[t * 2], x = cents[t * 2 + 1];
  size_t idx = (size_t)n * HWP + (size_t)y * WW + x;
  dist[idx] = 0.f;
  mask[idx] = ci;
}

// ------------- one direction of one propagation iteration ------------------
__global__ __launch_bounds__(256) void k_prop(const float* __restrict__ dsrc,
                                              const int* __restrict__ msrc,
                                              const float* __restrict__ pens,
                                              float* __restrict__ ddst,
                                              int* __restrict__ mdst, int d) {
  int p = blockIdx.x * 256 + threadIdx.x;
  if (p >= NPIX) return;
  int n = p / HWP, pi = p % HWP, y = pi / WW, xc = pi % WW;
  const int dys[4] = {-1, 1, 0, 0}, dxs[4] = {0, 0, -1, 1};
  int yy = (y - dys[d] + HH) % HH;
  int xx = (xc - dxs[d] + WW) % WW;
  size_t nb = (size_t)n * HWP + (size_t)yy * WW + xx;
  float wd = dsrc[nb] + pens[(size_t)d * NPIX + p];
  float dc = dsrc[p];
  if (wd < dc) { ddst[p] = wd; mdst[p] = msrc[nb]; }
  else         { ddst[p] = dc; mdst[p] = msrc[p]; }
}

__global__ __launch_bounds__(256) void k_copy_i32(const int* __restrict__ src,
                                                  int* __restrict__ dst) {
  int p = blockIdx.x * 256 + threadIdx.x;
  if (p >= NPIX) return;
  dst[p] = src[p];
}

// ---------------------------------------------------------------------------
extern "C" void kernel_launch(void* const* d_in, const int* in_sizes, int n_in,
                              void* d_out, int out_size, void* d_ws, size_t ws_size,
                              hipStream_t stream) {
  const float* x  = (const float*)d_in[0];
  const float* w1 = (const float*)d_in[1];
  const float* b1 = (const float*)d_in[2];
  const float* g1 = (const float*)d_in[3];
  const float* be1 = (const float*)d_in[4];
  const float* w2 = (const float*)d_in[5];
  const float* b2 = (const float*)d_in[6];   // cancels inside train-mode BN
  const float* g2 = (const float*)d_in[7];
  const float* be2 = (const float*)d_in[8];
  (void)b2; (void)in_sizes; (void)n_in; (void)out_size; (void)ws_size;

  // d_out layout (flat 4-byte elements): grad | cents(i32) | mask(i32) | feats
  float* out_grad = (float*)d_out;
  int*   out_cents = (int*)((char*)d_out + (size_t)NPIX * 4);
  int*   out_mask  = (int*)((char*)d_out + ((size_t)NPIX + BB * NCENT * 2) * 4);
  float* out_feats = (float*)((char*)d_out +
                     ((size_t)NPIX + BB * NCENT * 2 + NPIX) * 4);

  size_t off = 0;
  auto carve = [&](size_t bytes) {
    size_t r = off; off = (off + bytes + 255) & ~(size_t)255; return r;
  };
  char* ws = (char*)d_ws;
  float*    hpre  = (float*)(ws + carve((size_t)NPIX * C1 * 4));
  size_t    hfBytes = (size_t)BB * C1 * PLANE * 2;   // padded f16 activations
  _Float16* hf    = (_Float16*)(ws + carve(hfBytes));
  _Float16* wt    = (_Float16*)(ws + carve((size_t)KDIM * C2 * 2));
  float*    st1   = (float*)(ws + carve(C1 * 2 * 4));
  float*    st2   = (float*)(ws + carve(C2 * 2 * 4));
  float*    mf    = (float*)(ws + carve((size_t)NPIX * 4));
  float*    pens  = (float*)(ws + carve((size_t)4 * NPIX * 4));
  float*    distA = (float*)(ws + carve((size_t)NPIX * 4));
  float*    distB = (float*)(ws + carve((size_t)NPIX * 4));
  int*      maskA = (int*)(ws + carve((size_t)NPIX * 4));
  int*      maskB = (int*)(ws + carve((size_t)NPIX * 4));

  const int TPB = 256;
  const int gbNPIX = (NPIX + TPB - 1) / TPB;

  k_sobel<<<gbNPIX, TPB, 0, stream>>>(x, out_grad);

  k_conv1<<<gbNPIX, TPB, 0, stream>>>(x, w1, b1, hpre);
  hipMemsetAsync(st1, 0, C1 * 2 * 4, stream);
  hipMemsetAsync(hf, 0, hfBytes, stream);    // zero halo (f16 0.0 == 0x0000)
  k_bn_stats<<<dim3(64, C1), TPB, 0, stream>>>(hpre, st1, C1);
  {
    size_t tot = (size_t)NPIX * C1;
    k_bn1_f16<<<(unsigned)((tot + TPB - 1) / TPB), TPB, 0, stream>>>(
        hpre, st1, g1, be1, hf);
  }

  k_w2_f16<<<(C2 * KDIM + TPB - 1) / TPB, TPB, 0, stream>>>(w2, wt);
  k_conv2_wmma<<<NPIX / BM, TPB, 0, stream>>>(hf, wt, out_feats);

  hipMemsetAsync(st2, 0, C2 * 2 * 4, stream);
  k_bn_stats<<<dim3(64, C2), TPB, 0, stream>>>(out_feats, st2, C2);
  k_bn2_mean<<<gbNPIX, TPB, 0, stream>>>(out_feats, st2, g2, be2, mf);

  k_pens<<<gbNPIX, TPB, 0, stream>>>(x, out_grad, mf, pens);
  k_minima<<<BB, TPB, 0, stream>>>(out_grad, out_cents);
  k_init_dist<<<gbNPIX, TPB, 0, stream>>>(distA, maskA);
  k_seed<<<(BB * NCENT + TPB - 1) / TPB, TPB, 0, stream>>>(out_cents, distA, maskA);

  float* ds = distA; float* dd = distB;
  int* ms = maskA; int* md = maskB;
  for (int it = 0; it < 50; ++it) {
    for (int d = 0; d < 4; ++d) {
      k_prop<<<gbNPIX, TPB, 0, stream>>>(ds, ms, pens, dd, md, d);
      float* tf = ds; ds = dd; dd = tf;
      int* ti = ms; ms = md; md = ti;
    }
  }
  k_copy_i32<<<gbNPIX, TPB, 0, stream>>>(ms, out_mask);
}
